// MyRNN_6167573037381
// MI455X (gfx1250) — compile-verified, matrix operand order checked by reading, not run on hardware
//
#include <hip/hip_runtime.h>
#include <hip/hip_bf16.h>
#include <math.h>

// ---------------- problem constants ----------------
#define TSTEPS 512
#define BATCH  64
#define HID    1024
#define BH     (BATCH * HID)          // 65536
#define NWG    32                     // persistent workgroups
#define NBAR   (2 * TSTEPS)           // one barrier slot per layer-phase
#define LDS_BYTES (4 * 64 * 512 * 2)  // 4 chunks x 64 k-tiles x 512 bf16 = 256 KB

typedef __bf16 bf16;
typedef __attribute__((ext_vector_type(16))) __bf16 v16bf;
typedef __attribute__((ext_vector_type(8)))  __bf16 v8bf;
typedef __attribute__((ext_vector_type(8)))  float  v8f;
typedef __attribute__((ext_vector_type(4)))  unsigned int u32x4;
typedef __attribute__((ext_vector_type(4)))  float  f32x4;

// workspace layout (bytes)
#define OFF_BAR  0u
#define OFF_WB   4096u                                  // 2*64*64*512 bf16 = 8 MB (swizzled weights)
#define OFF_BIAS (OFF_WB + 2u*64u*64u*512u*2u)          // 2*1024 f32
#define OFF_H0   (OFF_BIAS + 2u*1024u*4u)               // 2 parities * BH bf16
#define OFF_H1   (OFF_H0 + 2u*BH*2u)
#define OFF_A1   (OFF_H1 + 2u*BH*2u)                    // BH bf16
#define OFF_XB   (OFF_A1 + BH*2u)                       // BH bf16 (staged x_t)

union BFrag  { v16bf v; u32x4 q[2]; };
union B8Frag { v8bf  v; u32x4 q;    };

// ---- A fragment (16x32 bf16, MxK) per ISA layout:
// lanes 0-15 : M=lane,   K = kt*32 + {0..7, 16..23}
// lanes 16-31: M=lane-16,K = kt*32 + {8..15, 24..31}
__device__ inline v16bf load_a(const bf16* __restrict__ base, int arow, int kt, int lane) {
    const bf16* p = base + arow * HID + kt * 32 + ((lane & 16) ? 8 : 0);
    BFrag f;
    f.q[0] = *reinterpret_cast<const u32x4*>(p);
    f.q[1] = *reinterpret_cast<const u32x4*>(p + 16);
    return f.v;
}

// ---- B fragment: pre-swizzled in LDS; each lane owns 16 contiguous bf16 (32B) per tile
__device__ inline v16bf load_b(const bf16* tile, int lane) {
    const bf16* p = tile + lane * 16;
    BFrag f;
    f.q[0] = *reinterpret_cast<const u32x4*>(p);
    f.q[1] = *reinterpret_cast<const u32x4*>(p + 8);
    return f.v;
}

__device__ inline v8f wmma_bf16(v16bf a, v16bf b, v8f c) {
    return __builtin_amdgcn_wmma_f32_16x16x32_bf16(false, a, false, b, (short)0, c, false, false);
}

// 32 k-tiles of K=32, two accumulators to break WMMA D->C dependency chains.
// A from global (L2-resident bf16), B from LDS.
__device__ inline void gemm_half(const bf16* __restrict__ aBase, const bf16* lwbL,
                                 int arow, int lane, v8f& c0, v8f& c1) {
#pragma unroll 8
    for (int kt = 0; kt < 32; kt += 2) {
        v16bf a0 = load_a(aBase, arow, kt, lane);
        v16bf b0 = load_b(lwbL + (size_t)kt * 512, lane);
        c0 = wmma_bf16(a0, b0, c0);
        v16bf a1 = load_a(aBase, arow, kt + 1, lane);
        v16bf b1 = load_b(lwbL + (size_t)(kt + 1) * 512, lane);
        c1 = wmma_bf16(a1, b1, c1);
    }
}

// device-wide barrier: arrive (release) + spin with s_sleep (acquire)
__device__ inline void grid_barrier(unsigned* __restrict__ bar, int slot) {
    __syncthreads();
    if (threadIdx.x == 0) {
        __hip_atomic_fetch_add(&bar[slot], 1u, __ATOMIC_RELEASE, __HIP_MEMORY_SCOPE_AGENT);
        while (__hip_atomic_load(&bar[slot], __ATOMIC_ACQUIRE, __HIP_MEMORY_SCOPE_AGENT) < (unsigned)NWG) {
            __builtin_amdgcn_s_sleep(2);
        }
    }
    __syncthreads();
}

// ---------------- prep kernels ----------------
__global__ void k_init_bar(unsigned* __restrict__ bar) {
    int i = blockIdx.x * blockDim.x + threadIdx.x;
    if (i < NBAR) bar[i] = 0u;
}

// Swizzle W' (k,n) with W'[k,n] = k<H ? Wx[l][n][k] : Wh[l][n][k-H] into
// per-lane WMMA B-fragment order: flat idx = (((l*64+ntile)*64+ktile)*32+lane)*16+j
__global__ void k_prep_weights(const float* __restrict__ Wx, const float* __restrict__ Wh,
                               bf16* __restrict__ wb) {
    int idx = blockIdx.x * blockDim.x + threadIdx.x;
    if (idx >= 2 * 64 * 64 * 32 * 16) return;
    int j     = idx & 15;
    int lane  = (idx >> 4) & 31;
    int ktile = (idx >> 9) & 63;
    int ntile = (idx >> 15) & 63;
    int l     = (idx >> 21) & 1;
    int kLoc  = (j & 7) + ((lane & 16) ? 8 : 0) + ((j & 8) ? 16 : 0);
    int k     = ktile * 32 + kLoc;
    int n     = ntile * 16 + (lane & 15);
    float v = (k < HID) ? Wx[(size_t)l * HID * HID + (size_t)n * HID + k]
                        : Wh[(size_t)l * HID * HID + (size_t)n * HID + (k - HID)];
    wb[idx] = (bf16)v;
}

__global__ void k_prep_misc(const float* __restrict__ bx, const float* __restrict__ bh,
                            const float* __restrict__ h0, const float* __restrict__ x,
                            float* __restrict__ bias,
                            bf16* __restrict__ h0buf, bf16* __restrict__ h1buf,
                            bf16* __restrict__ xbuf) {
    int i = blockIdx.x * blockDim.x + threadIdx.x;
    if (i < 2 * HID) bias[i] = bx[i] + bh[i];
    if (i < BH) {
        h0buf[BH + i] = (bf16)h0[i];        // parity 1 holds h_{-1} for layer 0
        h1buf[BH + i] = (bf16)h0[BH + i];   // parity 1 holds h_{-1} for layer 1
        xbuf[i]       = (bf16)x[i];         // x_0 staged for t=0
    }
}

// ---------------- persistent RNN kernel ----------------
__global__ __launch_bounds__(256) void k_rnn_persistent(
    const float* __restrict__ x,
    const bf16* __restrict__ wb, const float* __restrict__ bias,
    bf16* __restrict__ h0buf, bf16* __restrict__ h1buf, bf16* __restrict__ a1buf,
    bf16* __restrict__ xbuf,
    unsigned* __restrict__ bar, float* __restrict__ out) {

    extern __shared__ char smem[];
    bf16* lwb = (bf16*)smem;             // [4 chunks][64 k-tiles][512 halves]

    const int tid   = threadIdx.x;
    const int lane  = tid & 31;
    const int wave  = tid >> 5;          // 8 waves
    const int nSub  = wave >> 2;
    const int m0    = (wave & 3) * 16;   // batch tile
    const int nTile = blockIdx.x * 2 + nSub;
    const int n0    = nTile * 16;
    const int arow  = m0 + (lane & 15);              // A-fragment row (M)
    const int col   = n0 + (lane & 15);              // C/D column (N)
    const int rowB  = m0 + ((lane & 16) ? 8 : 0);    // C/D row base; +r per acc elem
    float* hn = out + (size_t)TSTEPS * BH;

    // ---- cache this WG's weight slice in LDS once (4 x 64KB contiguous chunks) ----
#pragma unroll
    for (int c = 0; c < 4; ++c) {
        int l = c >> 1;
        const u32x4* src = (const u32x4*)(wb + (size_t)(l * 64 + blockIdx.x * 2 + (c & 1)) * 64 * 512);
        u32x4* dst = (u32x4*)(lwb + (size_t)c * 64 * 512);
        for (int i = tid; i < 64 * 512 / 8; i += 256) dst[i] = src[i];
    }
    __syncthreads();

    const bf16* lb0 = lwb + (size_t)(0 * 2 + nSub) * 64 * 512;   // layer 0, this wave's n-subtile
    const bf16* lb1 = lwb + (size_t)(1 * 2 + nSub) * 64 * 512;   // layer 1
    const float b0v = bias[col];
    const float b1v = bias[HID + col];

    for (int t = 0; t < TSTEPS; ++t) {
        const int pr = (t + 1) & 1;   // parity holding h_{t-1}
        const int pw = t & 1;         // parity to write h_t

        // ================= layer 0 : A = [x_t | h0_{t-1}] =================
        {
            const bf16* hA = h0buf + pr * BH;
            v8f c0 = {0.f,0.f,0.f,0.f,0.f,0.f,0.f,0.f};
            v8f c1 = {0.f,0.f,0.f,0.f,0.f,0.f,0.f,0.f};
            gemm_half(xbuf, lb0,            arow, lane, c0, c1);   // K 0..1023   (x_t, staged bf16)
            gemm_half(hA,   lb0 + 32 * 512, arow, lane, c0, c1);   // K 1024..2047 (h0_{t-1})
#pragma unroll
            for (int r = 0; r < 8; ++r) {
                float h = tanhf(c0[r] + c1[r] + b0v);
                int row = rowB + r;
                bf16 hb = (bf16)h;
                a1buf[row * HID + col] = hb;                  // input to layer 1 (this step)
                h0buf[pw * BH + row * HID + col] = hb;        // recurrent input (next step)
                hn[row * HID + col] = h;                      // h_n layer 0 (final overwrite wins)
            }
        }
        grid_barrier(bar, 2 * t);

        // ================= layer 1 : A = [h0_t | h1_{t-1}] =================
        {
            const bf16* hA = h1buf + pr * BH;
            v8f c0 = {0.f,0.f,0.f,0.f,0.f,0.f,0.f,0.f};
            v8f c1 = {0.f,0.f,0.f,0.f,0.f,0.f,0.f,0.f};
            gemm_half(a1buf, lb1,            arow, lane, c0, c1);  // K 0..1023   (h0_t)
            gemm_half(hA,    lb1 + 32 * 512, arow, lane, c0, c1);  // K 1024..2047 (h1_{t-1})
#pragma unroll
            for (int r = 0; r < 8; ++r) {
                float h = tanhf(c0[r] + c1[r] + b1v);
                int row = rowB + r;
                out[(size_t)t * BH + row * HID + col] = 10.0f * h;
                h1buf[pw * BH + row * HID + col] = (bf16)h;
                hn[BH + row * HID + col] = h;
            }
            // ---- stage x_{t+1} -> bf16 once per step (8 elems/thread grid-wide);
            // readers of xbuf (layer-0 of t+1) are on the far side of barrier 2t+1,
            // and this step's readers finished before barrier 2t.
            if (t + 1 < TSTEPS) {
                const int gt = blockIdx.x * 256 + tid;               // 0..8191
                const float* xs = x + (size_t)(t + 1) * BH + (size_t)gt * 8;
                f32x4 u0 = *reinterpret_cast<const f32x4*>(xs);
                f32x4 u1 = *reinterpret_cast<const f32x4*>(xs + 4);
                B8Frag w;
                w.v[0]=(bf16)u0[0]; w.v[1]=(bf16)u0[1]; w.v[2]=(bf16)u0[2]; w.v[3]=(bf16)u0[3];
                w.v[4]=(bf16)u1[0]; w.v[5]=(bf16)u1[1]; w.v[6]=(bf16)u1[2]; w.v[7]=(bf16)u1[3];
                *reinterpret_cast<u32x4*>(xbuf + (size_t)gt * 8) = w.q;
            }
        }
        grid_barrier(bar, 2 * t + 1);
    }
}

// ---------------- launch ----------------
extern "C" void kernel_launch(void* const* d_in, const int* in_sizes, int n_in,
                              void* d_out, int out_size, void* d_ws, size_t ws_size,
                              hipStream_t stream) {
    const float* x  = (const float*)d_in[0];
    const float* h0 = (const float*)d_in[1];
    const float* Wx = (const float*)d_in[2];
    const float* bx = (const float*)d_in[3];
    const float* Wh = (const float*)d_in[4];
    const float* bh = (const float*)d_in[5];
    float* out = (float*)d_out;

    char* ws = (char*)d_ws;
    unsigned* bar  = (unsigned*)(ws + OFF_BAR);
    bf16* wb       = (bf16*)(ws + OFF_WB);
    float* bias    = (float*)(ws + OFF_BIAS);
    bf16* h0buf    = (bf16*)(ws + OFF_H0);
    bf16* h1buf    = (bf16*)(ws + OFF_H1);
    bf16* a1buf    = (bf16*)(ws + OFF_A1);
    bf16* xbuf     = (bf16*)(ws + OFF_XB);

    k_init_bar<<<4, 256, 0, stream>>>(bar);
    k_prep_weights<<<(2 * 64 * 64 * 32 * 16) / 256, 256, 0, stream>>>(Wx, Wh, wb);
    k_prep_misc<<<BH / 256, 256, 0, stream>>>(bx, bh, h0, x, bias, h0buf, h1buf, xbuf);
    k_rnn_persistent<<<NWG, 256, LDS_BYTES, stream>>>(x, wb, bias, h0buf, h1buf, a1buf,
                                                      xbuf, bar, out);
}
